// ModelAttention_46952582480553
// MI455X (gfx1250) — matrix operand order repaired and checked by reading, not compiled, not run on hardware
//
#include <hip/hip_runtime.h>
#include <math.h>

// ---------------------------------------------------------------------------
// Problem constants (match reference)
// ---------------------------------------------------------------------------
#define NN   30000
#define EE   360000
#define BB   8
#define FN   64
#define FE   16
#define DD   128
#define LL   3
#define MM   8
#define DKK  16
#define NEG_SLOPE 0.2f
#define EPSBN 1e-5f

typedef __attribute__((ext_vector_type(16))) _Float16 v16h;
typedef __attribute__((ext_vector_type(8)))  float    v8f;
typedef _Float16 half_t;

// Monotone float <-> uint key (for atomic max on floats)
__device__ __forceinline__ unsigned f2key(float f) {
  unsigned b = __float_as_uint(f);
  return (b & 0x80000000u) ? ~b : (b | 0x80000000u);
}
__device__ __forceinline__ float key2f(unsigned k) {
  unsigned b = (k & 0x80000000u) ? (k & 0x7FFFFFFFu) : ~k;
  return __uint_as_float(b);
}
#define NEGINF_KEY 0x007FFFFFu  // f2key(-inf)

// ---------------------------------------------------------------------------
// Weight packing: W [Dout=128, Kx] row-major (Y = X @ W^T) -> B-fragment order.
// Flat index = ((tn*nK + ks)*32 + lane)*16 + elem ; elem e of lane L holds
// B[k][n] with k = ks*32 + ((L&16)?16:0) + e, n = tn*16 + (L&15); B[k][n]=W[n][k].
// ---------------------------------------------------------------------------
__global__ void pack_w(const float* __restrict__ W, half_t* __restrict__ out,
                       int Kx, int Kp, int total) {
  int idx = blockIdx.x * 256 + threadIdx.x;
  if (idx >= total) return;
  int elem = idx & 15;
  int lane = (idx >> 4) & 31;
  int rest = idx >> 9;              // tn*nK + ks
  int nK   = Kp >> 5;
  int ks   = rest % nK;
  int tn   = rest / nK;
  int k    = ks * 32 + ((lane & 16) ? 16 : 0) + elem;
  int n    = tn * 16 + (lane & 15);
  float v  = (k < Kx) ? W[(size_t)n * Kx + k] : 0.0f;
  out[idx] = (half_t)v;
}

// ---------------------------------------------------------------------------
// Core per-wave 16x16 WMMA tile over a shared A tile (16 rows x Kp halves).
// ---------------------------------------------------------------------------
__device__ __forceinline__ v8f wmma_tile(const half_t* aT, int Kp,
                                         const half_t* Wp, int wave, int lane) {
  const int nK = Kp >> 5;
  v8f acc = {};
  const int m0     = lane & 15;
  const int kbaseA = (lane & 16) ? 16 : 0;
  const half_t* Bb = Wp + (size_t)wave * nK * 512;
  for (int ks = 0; ks < nK; ++ks) {
    v16h a, b;
    const int kk = ks * 32;
#pragma unroll
    for (int v = 0; v < 8; ++v) {
      int k0 = kk + kbaseA + 2 * v;
      a[2 * v]     = aT[m0 * Kp + k0];
      a[2 * v + 1] = aT[m0 * Kp + k0 + 1];
    }
    b = *(const v16h*)(Bb + (size_t)(ks * 32 + lane) * 16);
    acc = __builtin_amdgcn_wmma_f32_16x16x32_f16(false, a, false, b,
                                                 (short)0, acc, false, false);
  }
  return acc;
}

__device__ __forceinline__ void store_tile(v8f acc, float* __restrict__ Y,
                                           int rowBase, int R,
                                           const float* __restrict__ bias,
                                           int wave, int lane, int flags) {
  const int n    = wave * 16 + (lane & 15);
  const int mAdd = (lane >> 4) * 8;
  float bv = (flags & 1) ? bias[n] : 0.0f;
#pragma unroll
  for (int v = 0; v < 8; ++v) {
    int gr = rowBase + mAdd + v;
    if (gr < R) {
      float val = acc[v] + bv;
      if (flags & 2) val = fmaxf(val, 0.0f);
      Y[(size_t)gr * DD + n] = val;
    }
  }
}

// Y[R,128] = act(X[R,Kx] @ W^T + bias); Kp = Kx rounded up to 32.
__global__ __launch_bounds__(256)
void gemm_x128(const float* __restrict__ X, int Kx, int Kp,
               const half_t* __restrict__ Wp, const float* __restrict__ bias,
               float* __restrict__ Y, int R, int flags) {
  extern __shared__ half_t aT[];   // 16*Kp halves
  const int tid = threadIdx.x, wave = tid >> 5, lane = tid & 31;
  const int rowBase = blockIdx.x * 16;
  const int tot = 16 * Kp;         // multiple of 256 for all our Kp
  for (int idx = tid; idx < tot; idx += 256) {
    int r = idx / Kp, c = idx - r * Kp;
    int gr = rowBase + r;
    float v = (gr < R && c < Kx) ? X[(size_t)gr * Kx + c] : 0.0f;
    aT[idx] = (half_t)v;
  }
  __syncthreads();
  v8f acc = wmma_tile(aT, Kp, Wp, wave, lane);
  store_tile(acc, Y, rowBase, R, bias, wave, lane, flags);
}

// Final MLP hidden layer with fused concat-gather A loader:
// row e = [moy[edge_graph[e]] | h[src[e]] | h[dst[e]]]  (Kp = 384)
__global__ __launch_bounds__(256)
void gemm_concat384(const float* __restrict__ moy, const float* __restrict__ h,
                    const int* __restrict__ src, const int* __restrict__ dst,
                    const int* __restrict__ eg,
                    const half_t* __restrict__ Wp, const float* __restrict__ bias,
                    float* __restrict__ Y, int R) {
  __shared__ half_t aT[16 * 384];
  const int tid = threadIdx.x, wave = tid >> 5, lane = tid & 31;
  const int rowBase = blockIdx.x * 16;
  for (int idx = tid; idx < 16 * 384; idx += 256) {
    int r = idx / 384, c = idx - r * 384;
    int ge = rowBase + r;
    float v = 0.0f;
    if (ge < R) {
      if (c < 128)      v = moy[(size_t)eg[ge] * DD + c];
      else if (c < 256) v = h[(size_t)src[ge] * DD + (c - 128)];
      else              v = h[(size_t)dst[ge] * DD + (c - 256)];
    }
    aT[idx] = (half_t)v;
  }
  __syncthreads();
  v8f acc = wmma_tile(aT, 384, Wp, wave, lane);
  store_tile(acc, Y, rowBase, R, bias, wave, lane, 3 /* bias+relu */);
}

// ---------------------------------------------------------------------------
// Elementwise / gather / segment kernels
// ---------------------------------------------------------------------------
__global__ void fill_u32(unsigned* p, unsigned v, int n) {
  int i = blockIdx.x * 256 + threadIdx.x;
  if (i < n) p[i] = v;
}
__global__ void fill_f32(float* p, float v, int n) {
  int i = blockIdx.x * 256 + threadIdx.x;
  if (i < n) p[i] = v;
}

// m = Vh[src]*sigmoid(e); aggKey[dst] = max(aggKey[dst], key(m))   (fused)
__global__ void edge_mul_sigmoid_max(const float* __restrict__ Vh,
                                     const float* __restrict__ e,
                                     const int* __restrict__ src,
                                     const int* __restrict__ dst,
                                     unsigned* __restrict__ aggKey, int total) {
  int idx = blockIdx.x * 256 + threadIdx.x;
  if (idx >= total) return;
  int i = idx >> 7, d = idx & 127;
  float ev = e[idx];
  float m = Vh[(size_t)src[i] * DD + d] * (1.0f / (1.0f + expf(-ev)));
  atomicMax(&aggKey[(size_t)dst[i] * DD + d], f2key(m));
}

// out = Uh + decode(aggKey) (zero-in-degree -> 0)
__global__ void bn_input_nodes(float* __restrict__ out, const float* __restrict__ Uh,
                               const unsigned* __restrict__ aggKey, int total) {
  int idx = blockIdx.x * 256 + threadIdx.x;
  if (idx >= total) return;
  unsigned k = aggKey[idx];
  float a = (k == NEGINF_KEY) ? 0.0f : key2f(k);
  out[idx] = Uh[idx] + a;
}

// Column-wise sum / sumsq over [R,128] into stats[0..127]=sum, [128..255]=sumsq
__global__ __launch_bounds__(128)
void col_stats(const float* __restrict__ X, int R, float* __restrict__ stats) {
  const int c = threadIdx.x;           // 0..127
  const int r0 = blockIdx.x * 256;
  float s = 0.0f, s2 = 0.0f;
  int rEnd = r0 + 256; if (rEnd > R) rEnd = R;
  for (int r = r0; r < rEnd; ++r) {
    float v = X[(size_t)r * DD + c];
    s += v; s2 += v * v;
  }
  atomicAdd(&stats[c], s);
  atomicAdd(&stats[128 + c], s2);
}

// out = base + relu((x-mean)*rsqrt(var+eps))
__global__ void bn_apply_residual(const float* __restrict__ base,
                                  const float* __restrict__ x,
                                  const float* __restrict__ stats, float Rf,
                                  float* __restrict__ out, int total) {
  int idx = blockIdx.x * 256 + threadIdx.x;
  if (idx >= total) return;
  int d = idx & 127;
  float mean = stats[d] / Rf;
  float var  = stats[128 + d] / Rf - mean * mean;
  float v = (x[idx] - mean) * rsqrtf(var + EPSBN);
  out[idx] = base[idx] + fmaxf(v, 0.0f);
}

// t += Bh[dst] + Ch[src]
__global__ void edge_gather_add(float* __restrict__ t, const float* __restrict__ Bh,
                                const float* __restrict__ Ch,
                                const int* __restrict__ src,
                                const int* __restrict__ dst, int total) {
  int idx = blockIdx.x * 256 + threadIdx.x;
  if (idx >= total) return;
  int i = idx >> 7, d = idx & 127;
  t[idx] += Bh[(size_t)dst[i] * DD + d] + Ch[(size_t)src[i] * DD + d];
}

// GATv2 score + segment max  (idx over E*M)
__global__ void gat_score(const float* __restrict__ fs, const float* __restrict__ fd,
                          const int* __restrict__ src, const int* __restrict__ dst,
                          const float* __restrict__ attn,
                          float* __restrict__ score, unsigned* __restrict__ smaxKey,
                          int total) {
  int idx = blockIdx.x * 256 + threadIdx.x;
  if (idx >= total) return;
  int i = idx >> 3, hm = idx & 7;
  const float* ps = fs + (size_t)src[i] * DD + hm * DKK;
  const float* pd = fd + (size_t)dst[i] * DD + hm * DKK;
  float s = 0.0f;
#pragma unroll
  for (int k = 0; k < DKK; ++k) {
    float z = ps[k] + pd[k];
    z = (z > 0.0f) ? z : NEG_SLOPE * z;
    s += z * attn[hm * DKK + k];
  }
  score[idx] = s;
  atomicMax(&smaxKey[(size_t)dst[i] * MM + hm], f2key(s));
}

// p = exp(s - smax[dst]); denom[dst] += p
__global__ void gat_p(float* __restrict__ score, const unsigned* __restrict__ smaxKey,
                      const int* __restrict__ dst, float* __restrict__ denom, int total) {
  int idx = blockIdx.x * 256 + threadIdx.x;
  if (idx >= total) return;
  int i = idx >> 3, hm = idx & 7;
  float p = expf(score[idx] - key2f(smaxKey[(size_t)dst[i] * MM + hm]));
  score[idx] = p;
  atomicAdd(&denom[(size_t)dst[i] * MM + hm], p);
}

// rst[dst] += (p/denom[dst]) * fs[src]   (idx over E*128)
__global__ void gat_aggregate(const float* __restrict__ p, const float* __restrict__ denom,
                              const float* __restrict__ fs,
                              const int* __restrict__ src, const int* __restrict__ dst,
                              float* __restrict__ rst, int total) {
  int idx = blockIdx.x * 256 + threadIdx.x;
  if (idx >= total) return;
  int i = idx >> 7, d = idx & 127, hm = d >> 4;
  float a = p[(size_t)i * MM + hm] / denom[(size_t)dst[i] * MM + hm];
  atomicAdd(&rst[(size_t)dst[i] * DD + d], a * fs[(size_t)src[i] * DD + d]);
}

__global__ void add_row_bias(float* __restrict__ X, const float* __restrict__ bias, int total) {
  int idx = blockIdx.x * 256 + threadIdx.x;
  if (idx >= total) return;
  X[idx] += bias[idx & 127];
}

// Per-graph mean accumulation
__global__ void node_accum(const float* __restrict__ h, const int* __restrict__ ng,
                           float* __restrict__ moy, float* __restrict__ cnt, int total) {
  int idx = blockIdx.x * 256 + threadIdx.x;
  if (idx >= total) return;
  int n = idx >> 7, d = idx & 127;
  int g = ng[n];
  atomicAdd(&moy[(size_t)g * DD + d], h[idx]);
  if (d == 0) atomicAdd(&cnt[g], 1.0f);
}
__global__ void moy_scale(float* __restrict__ moy, const float* __restrict__ cnt, int total) {
  int idx = blockIdx.x * 256 + threadIdx.x;
  if (idx >= total) return;
  moy[idx] /= cnt[idx >> 7];
}

// out[e] = sigmoid(dot(hidden[e], W2) + b2)  — one wave32 per edge
__global__ void final_out(const float* __restrict__ H, const float* __restrict__ W2,
                          const float* __restrict__ b2, float* __restrict__ out, int E) {
  int gid  = blockIdx.x * blockDim.x + threadIdx.x;
  int lane = threadIdx.x & 31;
  int edge = gid >> 5;
  if (edge >= E) return;
  const float* row = H + (size_t)edge * DD;
  float s = 0.0f;
  for (int k = lane; k < DD; k += 32) s += row[k] * W2[k];
  for (int off = 16; off; off >>= 1) s += __shfl_xor(s, off, 32);
  if (lane == 0) out[edge] = 1.0f / (1.0f + expf(-(s + b2[0])));
}

// ---------------------------------------------------------------------------
// Host orchestration
// ---------------------------------------------------------------------------
extern "C" void kernel_launch(void* const* d_in, const int* in_sizes, int n_in,
                              void* d_out, int out_size, void* d_ws, size_t ws_size,
                              hipStream_t stream) {
  const float* h_raw  = (const float*)d_in[0];
  const float* e_raw  = (const float*)d_in[1];
  const int*   src    = (const int*)d_in[2];
  const int*   dst    = (const int*)d_in[3];
  const int*   ngph   = (const int*)d_in[4];
  const int*   egph   = (const int*)d_in[5];
  const float* emb_n_w = (const float*)d_in[6];
  const float* emb_n_b = (const float*)d_in[7];
  const float* emb_e_w = (const float*)d_in[8];
  const float* emb_e_b = (const float*)d_in[9];
  const float* Us = (const float*)d_in[10];
  const float* Vs = (const float*)d_in[11];
  const float* As = (const float*)d_in[12];
  const float* Bs = (const float*)d_in[13];
  const float* Cs = (const float*)d_in[14];
  const float* gat_src_w = (const float*)d_in[15];
  const float* gat_src_b = (const float*)d_in[16];
  const float* gat_dst_w = (const float*)d_in[17];
  const float* gat_dst_b = (const float*)d_in[18];
  const float* gat_attn  = (const float*)d_in[19];
  const float* gat_bias  = (const float*)d_in[20];
  const float* O_w  = (const float*)d_in[21];
  const float* O_b  = (const float*)d_in[22];
  const float* W1_w = (const float*)d_in[23];
  const float* W1_b = (const float*)d_in[24];
  const float* W2_w = (const float*)d_in[25];
  const float* W2_b = (const float*)d_in[26];

  // ---- workspace carving ----
  char* base = (char*)d_ws;
  size_t off = 0;
  auto carve = [&](size_t bytes) -> void* {
    void* p = base + off;
    off = (off + bytes + 255) & ~(size_t)255;
    return p;
  };
  const size_t ND = (size_t)NN * DD, ED = (size_t)EE * DD;
  float* hA  = (float*)carve(ND * 4);
  float* hB  = (float*)carve(ND * 4);
  float* tN1 = (float*)carve(ND * 4);
  float* tN2 = (float*)carve(ND * 4);
  float* tN3 = (float*)carve(ND * 4);
  float* eA  = (float*)carve(ED * 4);
  float* eB  = (float*)carve(ED * 4);
  float* tE1 = (float*)carve(ED * 4);
  float* sbuf = (float*)carve((size_t)EE * MM * 4);
  unsigned* smaxKey = (unsigned*)carve((size_t)NN * MM * 4);
  float* denom = (float*)carve((size_t)NN * MM * 4);
  float* moy = (float*)carve((size_t)BB * DD * 4);
  float* cnt = (float*)carve((size_t)BB * 4);
  float* stats = (float*)carve(256 * 4);
  auto carveW = [&](int Kx) -> half_t* {
    int Kp = (Kx + 31) & ~31;
    return (half_t*)carve((size_t)8 * (Kp >> 5) * 512 * sizeof(half_t));
  };
  half_t* pEmbN = carveW(FN);
  half_t* pEmbE = carveW(FE);
  half_t *pU[LL], *pV[LL], *pA[LL], *pB[LL], *pC[LL];
  for (int l = 0; l < LL; ++l) { pU[l]=carveW(DD); pV[l]=carveW(DD); pA[l]=carveW(DD); pB[l]=carveW(DD); pC[l]=carveW(DD); }
  half_t* pGS = carveW(DD);
  half_t* pGD = carveW(DD);
  half_t* pO  = carveW(DD);
  half_t* pW1 = carveW(3 * DD);

  auto pack = [&](const float* W, half_t* out, int Kx) {
    int Kp = (Kx + 31) & ~31;
    int total = 8 * (Kp >> 5) * 512;
    pack_w<<<(total + 255) / 256, 256, 0, stream>>>(W, out, Kx, Kp, total);
  };
  auto gemm = [&](const float* X, int Kx, const half_t* Wp, const float* bias,
                  float* Y, int R, int flags) {
    int Kp = (Kx + 31) & ~31;
    size_t lds = (size_t)16 * Kp * sizeof(half_t);
    gemm_x128<<<(R + 15) / 16, 256, lds, stream>>>(X, Kx, Kp, Wp, bias, Y, R, flags);
  };
  auto gridN = [](int total) { return dim3((unsigned)((total + 255) / 256)); };

  // ---- pack all weights (idempotent, deterministic) ----
  pack(emb_n_w, pEmbN, FN);
  pack(emb_e_w, pEmbE, FE);
  for (int l = 0; l < LL; ++l) {
    const size_t W = (size_t)DD * DD;
    pack(Us + l * W, pU[l], DD);
    pack(Vs + l * W, pV[l], DD);
    pack(As + l * W, pA[l], DD);
    pack(Bs + l * W, pB[l], DD);
    pack(Cs + l * W, pC[l], DD);
  }
  pack(gat_src_w, pGS, DD);
  pack(gat_dst_w, pGD, DD);
  pack(O_w, pO, DD);
  pack(W1_w, pW1, 3 * DD);

  // ---- embeddings ----
  gemm(h_raw, FN, pEmbN, emb_n_b, hA, NN, 1);
  gemm(e_raw, FE, pEmbE, emb_e_b, eA, EE, 1);

  float *hcur = hA, *hnxt = hB, *ecur = eA, *enxt = eB;
  const int NDtot = (int)ND, EDtot = (int)ED;

  // ---- L GatedGCN layers ----
  for (int l = 0; l < LL; ++l) {
    gemm(hcur, DD, pV[l], nullptr, tN1, NN, 0);                       // Vh
    unsigned* aggKey = (unsigned*)tN2;
    fill_u32<<<gridN(NDtot), 256, 0, stream>>>(aggKey, NEGINF_KEY, NDtot);
    edge_mul_sigmoid_max<<<gridN(EDtot), 256, 0, stream>>>(tN1, ecur, src, dst, aggKey, EDtot);
    gemm(hcur, DD, pU[l], nullptr, tN3, NN, 0);                       // Uh
    bn_input_nodes<<<gridN(NDtot), 256, 0, stream>>>(tN1, tN3, aggKey, NDtot);
    fill_f32<<<1, 256, 0, stream>>>(stats, 0.0f, 256);
    col_stats<<<(NN + 255) / 256, 128, 0, stream>>>(tN1, NN, stats);
    bn_apply_residual<<<gridN(NDtot), 256, 0, stream>>>(hcur, tN1, stats, (float)NN, hnxt, NDtot);
    // edge update (uses pre-update hcur)
    gemm(hcur, DD, pB[l], nullptr, tN2, NN, 0);                       // Bh
    gemm(hcur, DD, pC[l], nullptr, tN3, NN, 0);                       // Ch
    gemm(ecur, DD, pA[l], nullptr, tE1, EE, 0);                       // Ae
    edge_gather_add<<<gridN(EDtot), 256, 0, stream>>>(tE1, tN2, tN3, src, dst, EDtot);
    fill_f32<<<1, 256, 0, stream>>>(stats, 0.0f, 256);
    col_stats<<<(EE + 255) / 256, 128, 0, stream>>>(tE1, EE, stats);
    bn_apply_residual<<<gridN(EDtot), 256, 0, stream>>>(ecur, tE1, stats, (float)EE, enxt, EDtot);
    float* t;
    t = hcur; hcur = hnxt; hnxt = t;
    t = ecur; ecur = enxt; enxt = t;
  }

  // ---- GATv2 ----
  gemm(hcur, DD, pGS, gat_src_b, tN1, NN, 1);                         // fs
  gemm(hcur, DD, pGD, gat_dst_b, tN2, NN, 1);                         // fd
  const int EM = EE * MM, NM = NN * MM;
  fill_u32<<<gridN(NM), 256, 0, stream>>>(smaxKey, NEGINF_KEY, NM);
  gat_score<<<gridN(EM), 256, 0, stream>>>(tN1, tN2, src, dst, gat_attn, sbuf, smaxKey, EM);
  fill_f32<<<gridN(NM), 256, 0, stream>>>(denom, 0.0f, NM);
  gat_p<<<gridN(EM), 256, 0, stream>>>(sbuf, smaxKey, dst, denom, EM);
  fill_f32<<<gridN(NDtot), 256, 0, stream>>>(tN3, 0.0f, NDtot);
  gat_aggregate<<<gridN(EDtot), 256, 0, stream>>>(sbuf, denom, tN1, src, dst, tN3, EDtot);
  add_row_bias<<<gridN(NDtot), 256, 0, stream>>>(tN3, gat_bias, NDtot);
  gemm(tN3, DD, pO, O_b, hnxt, NN, 1);                                // final node states
  float* hF = hnxt;

  // ---- readout + edge MLP ----
  const int BD = BB * DD;
  fill_f32<<<1, 256, 0, stream>>>(moy, 0.0f, BD);
  fill_f32<<<1, 256, 0, stream>>>(cnt, 0.0f, BB);
  node_accum<<<gridN(NDtot), 256, 0, stream>>>(hF, ngph, moy, cnt, NDtot);
  moy_scale<<<1, 256, 0, stream>>>(moy, cnt, BD);
  gemm_concat384<<<(EE + 15) / 16, 256, 0, stream>>>(moy, hF, src, dst, egph,
                                                     pW1, W1_b, tE1, EE);
  final_out<<<(EE * 32 + 255) / 256, 256, 0, stream>>>(tE1, W2_w, W2_b, (float*)d_out, EE);
}
